// RAKT_17257178595532
// MI455X (gfx1250) — matrix operand order repaired and checked by reading, not compiled, not run on hardware
//
#include <hip/hip_runtime.h>
#include <hip/hip_bf16.h>

typedef __bf16 v16bf __attribute__((ext_vector_type(16)));
typedef float  v8f   __attribute__((ext_vector_type(8)));

#define B_     32
#define S_     512
#define E_     256
#define H_     8
#define Dh_    32
#define L_     2
#define NUMC_  200
#define M_     (B_ * S_)      /* 16384 rows */
#define MAXPOS_ 512

__device__ __forceinline__ v8f wmma_bf16(v16bf a, v16bf b, v8f c) {
  // D = A(16x32 bf16) * B(32x16 bf16) + C(16x16 f32)
  return __builtin_amdgcn_wmma_f32_16x16x32_bf16(false, a, false, b, (short)0, c,
                                                 false, false);
}

// ---------------------------------------------------------------------------
// Weight prep: fp32 [K,N] -> bf16 Bt [N,K]  (so B-fragment loads are contiguous)
// ---------------------------------------------------------------------------
__global__ void transpose_to_bf16(const float* __restrict__ src, __bf16* __restrict__ dst,
                                  int K, int N) {
  int i = blockIdx.x * blockDim.x + threadIdx.x;
  if (i >= K * N) return;
  int k = i / N, n = i - k * N;
  dst[(size_t)n * K + k] = (__bf16)src[i];
}

__global__ void convert_to_bf16(const float* __restrict__ src, __bf16* __restrict__ dst, int n) {
  int i = blockIdx.x * blockDim.x + threadIdx.x;
  if (i < n) dst[i] = (__bf16)src[i];
}

// ---------------------------------------------------------------------------
// Embedding gather: A512[row] = [ que_emb[pid] | mean_r concept_emb[c_data] ]
// (related = c+1 is never 0 -> count is always R=4)
// ---------------------------------------------------------------------------
__global__ __launch_bounds__(256) void gather_kernel(const int* __restrict__ pid,
                                                     const int* __restrict__ cdat,
                                                     const float* __restrict__ queE,
                                                     const float* __restrict__ conE,
                                                     float* __restrict__ A512) {
  int row = blockIdx.x;
  int e   = threadIdx.x;
  int p = pid[row];
  A512[(size_t)row * 512 + e] = queE[(size_t)p * E_ + e];
  const int* c = cdat + (size_t)row * 4;
  float s = conE[(size_t)c[0] * E_ + e] + conE[(size_t)c[1] * E_ + e] +
            conE[(size_t)c[2] * E_ + e] + conE[(size_t)c[3] * E_ + e];
  A512[(size_t)row * 512 + 256 + e] = 0.25f * s;
}

// inter input: [ query*lbl | query*(1-lbl) ]
__global__ __launch_bounds__(256) void build_inter_A(const float* __restrict__ query,
                                                     const int* __restrict__ target,
                                                     float* __restrict__ A512) {
  int row = blockIdx.x;
  int e   = threadIdx.x;
  float l = (float)target[row];
  float q = query[(size_t)row * E_ + e];
  A512[(size_t)row * 512 + e]       = q * l;
  A512[(size_t)row * 512 + 256 + e] = q * (1.0f - l);
}

// h1 input: [ outputs | query ]
__global__ __launch_bounds__(256) void build_concat(const float* __restrict__ X,
                                                    const float* __restrict__ query,
                                                    float* __restrict__ A512) {
  int row = blockIdx.x;
  int e   = threadIdx.x;
  A512[(size_t)row * 512 + e]       = X[(size_t)row * E_ + e];
  A512[(size_t)row * 512 + 256 + e] = query[(size_t)row * E_ + e];
}

__global__ void add_relu(float* __restrict__ X, const float* __restrict__ R, int n) {
  int i = blockIdx.x * blockDim.x + threadIdx.x;
  if (i < n) X[i] += fmaxf(R[i], 0.0f);
}

// ---------------------------------------------------------------------------
// WMMA GEMM: C[M,N] = act(A[M,K](f32) * Bt[N,K](bf16)^T + bias)
// 4 waves per block: one 16-col N tile, 64 rows (16 per wave). The Bt tile
// (16 x K bf16) is staged into LDS ONCE per block with CDNA5 async copies
// (global_load_async_to_lds_b128 + s_wait_asynccnt), then all 4 waves read
// their B fragments from LDS -> 4x less B fetch, vmem freed for A streams.
// act: 0 = none, 1 = relu, 2 = sigmoid
// ---------------------------------------------------------------------------
__global__ __launch_bounds__(128) void gemm_bf16_kernel(const float* __restrict__ A, int lda,
                                                        const __bf16* __restrict__ Bt,
                                                        const float* __restrict__ bias,
                                                        float* __restrict__ C, int ldc,
                                                        int N, int K, int act) {
  extern __shared__ char bsm_raw[];
  __bf16* Bs = (__bf16*)bsm_raw;                 // [16][K]

  const int tid  = threadIdx.x;                  // 0..127
  const int lane = tid & 31;
  const int wid  = tid >> 5;                     // wave -> M sub-tile
  const int g  = lane >> 4;
  const int mn = lane & 15;
  const int n0 = blockIdx.x * 16;
  const int m0 = blockIdx.y * 64 + wid * 16;

  // ---- async stage of Bt[n0 .. n0+15][0..K) into LDS (16B per lane/issue) ----
  {
    const char* src0 = (const char*)(Bt + (size_t)n0 * K);
    const int totalBytes = 16 * K * 2;
    for (int off = tid * 16; off < totalBytes; off += 128 * 16) {
      unsigned ldsoff = (unsigned)off;
      unsigned long long gaddr = (unsigned long long)(src0 + off);
      asm volatile("global_load_async_to_lds_b128 %0, %1, off"
                   :: "v"(ldsoff), "v"(gaddr) : "memory");
    }
    asm volatile("s_wait_asynccnt 0x0" ::: "memory");
    __syncthreads();
  }

  const int n = n0 + mn;
  const float*  arow = A + (size_t)(m0 + mn) * lda;
  const __bf16* brow = Bs + (size_t)mn * K + (g << 4);   // LDS

  v8f acc = {};
  for (int k0 = 0; k0 < K; k0 += 32) {
    __builtin_prefetch(arow + k0 + 64, 0, 1);            // global_prefetch_b8
    v16bf a, b;
#pragma unroll
    for (int t = 0; t < 16; ++t) {
      int ka = (g << 4) + t;                             // contiguous k = 16*g + t (matches B)
      a[t] = (__bf16)arow[k0 + ka];
    }
#pragma unroll
    for (int t = 0; t < 16; ++t) b[t] = brow[k0 + t];    // contiguous k = 16*g + t (matches A)
    acc = wmma_bf16(a, b, acc);
  }

  const int nc = (n < N) ? n : (N - 1);
  float bvv = bias ? bias[nc] : 0.0f;
#pragma unroll
  for (int t = 0; t < 8; ++t) {
    int m = m0 + t + (g << 3);
    float v = acc[t] + bvv;
    if (act == 1) v = fmaxf(v, 0.0f);
    else if (act == 2) v = 1.0f / (1.0f + __expf(-v));
    if (n < N) C[(size_t)m * ldc + n] = v;
  }
}

// ---------------------------------------------------------------------------
// Fused relative attention, one wave per (b, h, 16-row i-tile).
//   scores = Q Kt + scatter(Q PKt); strict-causal softmax (scaled on read);
//   out    = P V + PS PV      (PS[i,r] = P[i,i-r], scatter done in softmax)
// QP tiles scatter-add straight into the score rows: C-frag element (m, r)
// maps to the unique column j = i0+m-r, so no separate QP buffer is needed.
// LDS: 16x512 f32 scores + 2 x 16x544 bf16 (P, PS) = 66 KB; no HBM S*S tensor.
// ---------------------------------------------------------------------------
#define PROWS 544                                /* padded bf16 row stride */
#define ATT_SMEM (16 * 512 * 4 + 16 * PROWS * 2 * 2)

__global__ __launch_bounds__(32) void rel_attn_kernel(const float* __restrict__ Q,
                                                      const float* __restrict__ Km,
                                                      const float* __restrict__ V,
                                                      const __bf16* __restrict__ PK,  // [512,32]
                                                      const __bf16* __restrict__ PVt, // [32,512]
                                                      float* __restrict__ O) {
  extern __shared__ char smem_raw[];
  float*  sc = (float*)smem_raw;                    // [16][512] raw scores -> exp
  __bf16* pb = (__bf16*)(sc + 16 * 512);            // [16][PROWS] P   (bf16)
  __bf16* ps = pb + 16 * PROWS;                     // [16][PROWS] PS  (bf16)

  const int lane = threadIdx.x;
  const int g  = lane >> 4;
  const int mn = lane & 15;
  const int i0 = blockIdx.x * 16;
  const int h  = blockIdx.y;
  const int b  = blockIdx.z;
  const size_t base = (size_t)b * S_ * E_ + (size_t)h * Dh_;

  for (int idx = lane; idx < 16 * PROWS; idx += 32) {
    pb[idx] = (__bf16)0.0f;
    ps[idx] = (__bf16)0.0f;
  }

  // Q fragment: K-dim is exactly Dh=32 -> a single WMMA K-step, reused.
  v16bf aq;
  {
    const float* qrow = Q + base + (size_t)(i0 + mn) * E_;
#pragma unroll
    for (int t = 0; t < 16; ++t) {
      int ka = (g << 4) + t;
      aq[t] = (__bf16)qrow[ka];
    }
  }

  const int jtiles = (i0 >> 4) + 1;                 // j0 = 0..i0 step 16
  for (int jt = 0; jt < jtiles; ++jt) {
    int j0 = jt * 16;
    v16bf bk;
    const float* krow = Km + base + (size_t)(j0 + mn) * E_ + (g << 4);
#pragma unroll
    for (int t = 0; t < 16; ++t) bk[t] = (__bf16)krow[t];
    v8f c = {};
    c = wmma_bf16(aq, bk, c);
#pragma unroll
    for (int t = 0; t < 8; ++t) sc[(t + (g << 3)) * 512 + j0 + mn] = c[t];
  }
  for (int rt = 0; rt < jtiles; ++rt) {             // QP tile: r0 = 0..i0
    int r0 = rt * 16;
    v16bf bp;
    const __bf16* pkrow = PK + (size_t)(r0 + mn) * Dh_ + (g << 4);
#pragma unroll
    for (int t = 0; t < 16; ++t) bp[t] = pkrow[t];
    v8f c = {};
    c = wmma_bf16(aq, bp, c);
#pragma unroll
    for (int t = 0; t < 8; ++t) {                   // scatter-add: j = i - r
      int m = t + (g << 3);
      int j = i0 + m - (r0 + mn);
      if (j >= 0) sc[m * 512 + j] += c[t];          // unique (m,j) per element
    }
  }
  __syncthreads();

  // Row-wise strict-causal softmax; scatter PS[i, i-j] while normalizing.
  const float scale = 0.17677669529663687f;         // 1/sqrt(32)
  for (int m = 0; m < 16; ++m) {
    int i = i0 + m;
    if (i == 0) continue;                           // uniform branch; row stays 0
    float mx = -1e30f;
    for (int j = lane; j < i; j += 32)
      mx = fmaxf(mx, sc[m * 512 + j] * scale);
#pragma unroll
    for (int o = 16; o; o >>= 1) mx = fmaxf(mx, __shfl_xor(mx, o, 32));
    float sum = 0.0f;
    for (int j = lane; j < i; j += 32) {
      float e = __expf(sc[m * 512 + j] * scale - mx);
      sc[m * 512 + j] = e;
      sum += e;
    }
#pragma unroll
    for (int o = 16; o; o >>= 1) sum += __shfl_xor(sum, o, 32);
    float inv = 1.0f / sum;
    for (int j = lane; j < i; j += 32) {
      float p = sc[m * 512 + j] * inv;
      pb[m * PROWS + j]       = (__bf16)p;
      ps[m * PROWS + (i - j)] = (__bf16)p;
    }
  }
  __syncthreads();

  // out = P V + PS PV   (N = Dh = 32 -> two 16-wide N tiles)
  const int ktiles = (i0 + 16 + 31) >> 5;           // K span covers j < i0+16
  for (int n0 = 0; n0 < Dh_; n0 += 16) {
    v8f acc = {};
    const float*  vcol  = V + base + n0 + mn;
    const __bf16* pvrow = PVt + (size_t)(n0 + mn) * MAXPOS_;
    for (int kt = 0; kt < ktiles; ++kt) {
      int k0 = kt * 32;
      int kb = k0 + (g << 4);
      v16bf ap, bv, aps, bpv;
#pragma unroll
      for (int t = 0; t < 16; ++t) {
        int ka = k0 + (g << 4) + t;
        ap[t]  = pb[mn * PROWS + ka];
        aps[t] = ps[mn * PROWS + ka];
      }
#pragma unroll
      for (int t = 0; t < 16; ++t) {
        int j = kb + t;
        bv[t]  = (j < S_) ? (__bf16)vcol[(size_t)j * E_] : (__bf16)0.0f;
        bpv[t] = (j < MAXPOS_) ? pvrow[j] : (__bf16)0.0f;
      }
      acc = wmma_bf16(ap, bv, acc);
      acc = wmma_bf16(aps, bpv, acc);
    }
#pragma unroll
    for (int t = 0; t < 8; ++t)
      O[base + (size_t)(i0 + t + (g << 3)) * E_ + n0 + mn] = acc[t];
  }
}

// ---------------------------------------------------------------------------
// Final head: pred = sigmoid(h2 . o3_W + o3_b); one wave per row.
// ---------------------------------------------------------------------------
__global__ __launch_bounds__(256) void final_kernel(const float* __restrict__ h2,
                                                    const float* __restrict__ o3W,
                                                    const float* __restrict__ o3b,
                                                    float* __restrict__ out) {
  int row  = blockIdx.x * 8 + (threadIdx.x >> 5);
  int lane = threadIdx.x & 31;
  if (row >= M_) return;
  float s = 0.0f;
  for (int e = lane; e < E_; e += 32) s += h2[(size_t)row * E_ + e] * o3W[e];
#pragma unroll
  for (int o = 16; o; o >>= 1) s += __shfl_xor(s, o, 32);
  if (lane == 0) out[row] = 1.0f / (1.0f + __expf(-(s + o3b[0])));
}

// ---------------------------------------------------------------------------
extern "C" void kernel_launch(void* const* d_in, const int* in_sizes, int n_in,
                              void* d_out, int out_size, void* d_ws, size_t ws_size,
                              hipStream_t stream) {
  const int*   pid  = (const int*)d_in[0];
  const int*   cdat = (const int*)d_in[1];
  const int*   targ = (const int*)d_in[2];
  const float* queE = (const float*)d_in[3];
  const float* conE = (const float*)d_in[4];
  const float* posK = (const float*)d_in[5];
  const float* posV = (const float*)d_in[6];
  const float* qcW  = (const float*)d_in[7];
  const float* qcb  = (const float*)d_in[8];
  const float* linW = (const float*)d_in[9];
  const float* linb = (const float*)d_in[10];
  const float* Wq   = (const float*)d_in[11];
  const float* bq   = (const float*)d_in[12];
  const float* Wk   = (const float*)d_in[13];
  const float* bk   = (const float*)d_in[14];
  const float* Wv   = (const float*)d_in[15];
  const float* bv   = (const float*)d_in[16];
  const float* o1W  = (const float*)d_in[17];
  const float* o1b  = (const float*)d_in[18];
  const float* o2W  = (const float*)d_in[19];
  const float* o2b  = (const float*)d_in[20];
  const float* o3W  = (const float*)d_in[21];
  const float* o3b  = (const float*)d_in[22];

  float* outPred = (float*)d_out;            // [B,S]
  float* outSim  = (float*)d_out + M_;       // [B,S,NUMC]

  // ---- workspace carve-up (f32 then bf16) ----
  float* A512 = (float*)d_ws;                        // [M,512] (reused 3x)
  float* Qry  = A512 + (size_t)M_ * 512;             // [M,E]
  float* Itr  = Qry  + (size_t)M_ * E_;
  float* Xb   = Itr  + (size_t)M_ * E_;
  float* Rb   = Xb   + (size_t)M_ * E_;
  float* Qb   = Rb   + (size_t)M_ * E_;
  float* Kb   = Qb   + (size_t)M_ * E_;
  float* Vb   = Kb   + (size_t)M_ * E_;
  float* H1   = Vb   + (size_t)M_ * E_;
  float* H2   = H1   + (size_t)M_ * E_;
  __bf16* qcWt = (__bf16*)(H2 + (size_t)M_ * E_);
  __bf16* linWt = qcWt  + 512 * E_;
  __bf16* WqT   = linWt + 512 * E_;                  // [L][E,E] transposed
  __bf16* WkT   = WqT   + L_ * E_ * E_;
  __bf16* WvT   = WkT   + L_ * E_ * E_;
  __bf16* o1t   = WvT   + L_ * E_ * E_;
  __bf16* o2t   = o1t   + 512 * E_;
  __bf16* CEb   = o2t   + E_ * E_;
  __bf16* PKb   = CEb   + NUMC_ * E_;
  __bf16* PVt   = PKb   + MAXPOS_ * Dh_;

  auto blocks = [](int n) { return (n + 255) / 256; };

  // ---- weight prep ----
  transpose_to_bf16<<<blocks(512 * E_), 256, 0, stream>>>(qcW, qcWt, 512, E_);
  transpose_to_bf16<<<blocks(512 * E_), 256, 0, stream>>>(linW, linWt, 512, E_);
  for (int l = 0; l < L_; ++l) {
    transpose_to_bf16<<<blocks(E_ * E_), 256, 0, stream>>>(Wq + (size_t)l * E_ * E_, WqT + (size_t)l * E_ * E_, E_, E_);
    transpose_to_bf16<<<blocks(E_ * E_), 256, 0, stream>>>(Wk + (size_t)l * E_ * E_, WkT + (size_t)l * E_ * E_, E_, E_);
    transpose_to_bf16<<<blocks(E_ * E_), 256, 0, stream>>>(Wv + (size_t)l * E_ * E_, WvT + (size_t)l * E_ * E_, E_, E_);
  }
  transpose_to_bf16<<<blocks(512 * E_), 256, 0, stream>>>(o1W, o1t, 512, E_);
  transpose_to_bf16<<<blocks(E_ * E_), 256, 0, stream>>>(o2W, o2t, E_, E_);
  convert_to_bf16<<<blocks(NUMC_ * E_), 256, 0, stream>>>(conE, CEb, NUMC_ * E_);
  convert_to_bf16<<<blocks(MAXPOS_ * Dh_), 256, 0, stream>>>(posK, PKb, MAXPOS_ * Dh_);
  transpose_to_bf16<<<blocks(MAXPOS_ * Dh_), 256, 0, stream>>>(posV, PVt, MAXPOS_, Dh_);

  auto gemm = [&](const float* A, int lda, const __bf16* Bt, const float* bias,
                  float* C, int ldc, int N, int K, int act) {
    dim3 grid((N + 15) / 16, M_ / 64);
    gemm_bf16_kernel<<<grid, 128, (size_t)32 * K, stream>>>(A, lda, Bt, bias, C, ldc, N, K, act);
  };

  // ---- embeddings + query + sim + interaction ----
  gather_kernel<<<M_, 256, 0, stream>>>(pid, cdat, queE, conE, A512);
  gemm(A512, 512, CEb, nullptr, outSim, NUMC_, NUMC_, E_, 2);   // sim = sigmoid(qemb.CE^T)
  gemm(A512, 512, qcWt, qcb, Qry, E_, E_, 512, 0);              // query
  build_inter_A<<<M_, 256, 0, stream>>>(Qry, targ, A512);
  gemm(A512, 512, linWt, linb, Itr, E_, E_, 512, 1);            // inter = relu(...)

  // ---- attention layers ----
  dim3 ag(S_ / 16, H_, B_);
  // layer 0: q,k from query; v from inter -> Xb
  gemm(Qry, E_, WqT, bq, Qb, E_, E_, E_, 0);
  gemm(Qry, E_, WkT, bk, Kb, E_, E_, E_, 0);
  gemm(Itr, E_, WvT, bv, Vb, E_, E_, E_, 0);
  rel_attn_kernel<<<ag, 32, ATT_SMEM, stream>>>(Qb, Kb, Vb, PKb, PVt, Xb);
  // layer 1: q,k from Xb; v from inter -> Rb; Xb += relu(Rb)
  gemm(Xb, E_, WqT + (size_t)E_ * E_, bq + E_, Qb, E_, E_, E_, 0);
  gemm(Xb, E_, WkT + (size_t)E_ * E_, bk + E_, Kb, E_, E_, E_, 0);
  gemm(Itr, E_, WvT + (size_t)E_ * E_, bv + E_, Vb, E_, E_, E_, 0);
  rel_attn_kernel<<<ag, 32, ATT_SMEM, stream>>>(Qb, Kb, Vb, PKb, PVt, Rb);
  add_relu<<<blocks(M_ * E_), 256, 0, stream>>>(Xb, Rb, M_ * E_);

  // ---- output MLP ----
  build_concat<<<M_, 256, 0, stream>>>(Xb, Qry, A512);
  gemm(A512, 512, o1t, o1b, H1, E_, E_, 512, 1);
  gemm(H1, E_, o2t, o2b, H2, E_, E_, E_, 1);
  final_kernel<<<(M_ + 7) / 8, 256, 0, stream>>>(H2, o3W, o3b, outPred);
}
